// MPNN_75445395521649
// MI455X (gfx1250) — matrix operand order repaired and checked by reading, not compiled, not run on hardware
//
#include <hip/hip_runtime.h>

#define N_NODES  16384
#define N_EDGES  65536
#define N_GRAPHS 64

typedef __attribute__((ext_vector_type(16))) __bf16 v16bf;
typedef __attribute__((ext_vector_type(8)))  float  v8f;

// ---------------------------------------------------------------------------
// h[e,k] = prelu(attr[e]*w1[k] + b1[k]); stored bf16, per-row swizzled into
// WMMA 16-bit A order: pos = c*32 + g*16 + j  ->  K = c*32 + g*8 + j + (j<8?0:8)
// (lane group g in {0,1}; c = K-chunk of 32)
// ---------------------------------------------------------------------------
__global__ void k_edge_h(const float* __restrict__ attr,
                         const float* __restrict__ w1,
                         const float* __restrict__ b1,
                         const float* __restrict__ a,
                         __bf16* __restrict__ hsw) {
  int t = blockIdx.x * blockDim.x + threadIdx.x;
  if (t >= N_EDGES * 64) return;
  int e = t >> 6, p = t & 63;
  int c = p >> 5, rem = p & 31, g = rem >> 4, j = rem & 15;
  int k = c * 32 + g * 8 + j + (j < 8 ? 0 : 8);
  float v = attr[e] * w1[k] + b1[k];
  v = v >= 0.f ? v : a[0] * v;
  hsw[t] = (__bf16)v;
}

// ---------------------------------------------------------------------------
// Repack w2 [in*64, 64] into bf16 B' in WMMA 16-bit B lane order:
// linear t = (((C32*4 + ob)*2 + g)*16 + n)*16 + j
//   K = C32*32 + g*8 + j + (j<8?0:8) ; o = ob*16 + n ; value = w2[(i*64+o)*64 + k],
//   i = K/64, k = K%64.
// ---------------------------------------------------------------------------
__global__ void k_pack_w2(const float* __restrict__ w2,
                          __bf16* __restrict__ pB, int in_ch) {
  int t = blockIdx.x * blockDim.x + threadIdx.x;
  int total = in_ch * 64 * 64;
  if (t >= total) return;
  int j = t & 15, n = (t >> 4) & 15, g = (t >> 8) & 1, ob = (t >> 9) & 3;
  int C32 = t >> 11;
  int K = C32 * 32 + g * 8 + j + (j < 8 ? 0 : 8);
  int o = ob * 16 + n;
  int i = K >> 6, k = K & 63;
  pB[t] = (__bf16)w2[(i * 64 + o) * 64 + k];
}

// ---------------------------------------------------------------------------
// R[n,o] = bias[o] + sum_i x[n,i]*root[i,o];  B2x[n,o] = sum_i x[n,i]*b2[i*64+o]
// ---------------------------------------------------------------------------
__global__ void k_xroot(const float* __restrict__ x, const float* __restrict__ root,
                        const float* __restrict__ bias, const float* __restrict__ b2,
                        float* __restrict__ R, float* __restrict__ B2x, int in_ch) {
  int t = blockIdx.x * blockDim.x + threadIdx.x;
  if (t >= N_NODES * 64) return;
  int n = t >> 6, o = t & 63;
  float r = bias[o], s = 0.f;
  for (int i = 0; i < in_ch; ++i) {
    float xv = x[n * in_ch + i];
    r += xv * root[i * 64 + o];
    s += xv * b2[i * 64 + o];
  }
  R[t] = r;
  B2x[t] = s;
}

// ---------------------------------------------------------------------------
// Main NNConv as one fused GEMM:  msg = Z @ W2',  Z[e, i*64+k] = x[src_e,i]*h[e,k]
// 8 waves per WG; each wave owns a 16-edge tile and a 16x64 f32 accumulator.
// B' staged via LDS in 256-K superchunks; A built in registers, double-buffered
// so the WMMA->VALU WAR hazard on the A registers is covered by the other
// buffer's 4 WMMAs (no v_nop insertion needed).
// Epilogue: += x[src]@b2 term, atomic scatter-add into agg[dst].
// ---------------------------------------------------------------------------
__global__ void __launch_bounds__(256) k_nnconv(
    const float*  __restrict__ x,
    const __bf16* __restrict__ hsw,
    const __bf16* __restrict__ pB,
    const int*    __restrict__ src,
    const int*    __restrict__ dst,
    const float*  __restrict__ B2x,
    float*        __restrict__ agg,
    int in_ch) {
  __shared__ __bf16 sB[8 * 2048];     // 32 KB: one 256-K superchunk of B'
  __shared__ __bf16 sX[8 * 16 * 64];  // 16 KB: x rows of 16 src nodes per wave
  __shared__ int    sI[8 * 32];       // src[16], dst[16] per wave

  const int tid  = threadIdx.x;
  const int lane = tid & 31;
  const int wv   = tid >> 5;
  const int m    = lane & 15;  // A row / C column index
  const int g    = lane >> 4;  // lane group (K-half for A/B, row-half for C)
  const int e0   = (blockIdx.x * 8 + wv) * 16;

  if (lane < 16) {
    sI[wv * 32 + lane]      = src[e0 + lane];
    sI[wv * 32 + 16 + lane] = dst[e0 + lane];
  }
  for (int idx = lane; idx < 16 * in_ch; idx += 32) {
    int row = idx / in_ch, col = idx - row * in_ch;
    int s = sI[wv * 32 + row];
    sX[(wv * 16 + row) * 64 + col] = (__bf16)x[s * in_ch + col];
  }

  // h row for this lane, both 32-K halves, already in WMMA-A order
  const __bf16* hrow = hsw + (size_t)(e0 + m) * 64 + g * 16;
  const v16bf hv0 = *(const v16bf*)(hrow);
  const v16bf hv1 = *(const v16bf*)(hrow + 32);

  v8f acc[4] = {};
  const int nsc = (in_ch * 64) >> 8;  // superchunks of 256 K
  for (int sc = 0; sc < nsc; ++sc) {
    __syncthreads();
    const float4* gs = (const float4*)(pB + (size_t)sc * 16384);
    float4* ls = (float4*)sB;
#pragma unroll
    for (int q = 0; q < 8; ++q) ls[tid + q * 256] = gs[tid + q * 256];
    __syncthreads();

    // A builder for chunk c of this superchunk (32 K-values)
    auto buildA = [&](int c) -> v16bf {
      const float xs = (float)sX[(wv * 16 + m) * 64 + ((sc * 4) + (c >> 1))];
      const v16bf hv = (c & 1) ? hv1 : hv0;
      v16bf av;
#pragma unroll
      for (int j = 0; j < 16; ++j) av[j] = (__bf16)(xs * (float)hv[j]);
      return av;
    };

    v16bf av0 = buildA(0);
    v16bf av1 = buildA(1);
#pragma unroll
    for (int c = 0; c < 8; c += 2) {
      const __bf16* bb = sB + c * 2048 + g * 256 + m * 16;
#pragma unroll
      for (int ob = 0; ob < 4; ++ob) {
        v16bf bv = *(const v16bf*)(bb + ob * 512);
        acc[ob] = __builtin_amdgcn_wmma_f32_16x16x32_bf16(
            false, av0, false, bv, (short)0, acc[ob], false, false);
      }
#pragma unroll
      for (int ob = 0; ob < 4; ++ob) {
        v16bf bv = *(const v16bf*)(bb + 2048 + ob * 512);
        acc[ob] = __builtin_amdgcn_wmma_f32_16x16x32_bf16(
            false, av1, false, bv, (short)0, acc[ob], false, false);
      }
      if (c + 2 < 8) {  // rebuild A only after 8 intervening WMMAs
        av0 = buildA(c + 2);
        av1 = buildA(c + 3);
      }
    }
  }

  // C layout: VGPR r, lanes 0-15 -> M=r, N=lane ; lanes 16-31 -> M=r+8, N=lane-16
#pragma unroll
  for (int r = 0; r < 8; ++r) {
    const int er = r + 8 * g;
    const int s = sI[wv * 32 + er];
    const int d = sI[wv * 32 + 16 + er];
#pragma unroll
    for (int ob = 0; ob < 4; ++ob) {
      const int o = ob * 16 + m;
      atomicAdd(&agg[(size_t)d * 64 + o], acc[ob][r] + B2x[(size_t)s * 64 + o]);
    }
  }
}

// ---------------------------------------------------------------------------
__global__ void k_finalize(const float* __restrict__ agg, const float* __restrict__ R,
                           const float* __restrict__ pa, float* __restrict__ xout) {
  int t = blockIdx.x * blockDim.x + threadIdx.x;
  if (t >= N_NODES * 64) return;
  float v = agg[t] + R[t];
  xout[t] = v >= 0.f ? v : pa[0] * v;
}

__global__ void k_pool(const float* __restrict__ xf, const int* __restrict__ batch,
                       const float* __restrict__ ow, float* __restrict__ out) {
  int n = blockIdx.x * blockDim.x + threadIdx.x;
  if (n >= N_NODES) return;
  float s = 0.f;
#pragma unroll
  for (int o = 0; o < 64; ++o) s += xf[(size_t)n * 64 + o] * ow[o];
  atomicAdd(&out[batch[n]], s);
}

// ---------------------------------------------------------------------------
extern "C" void kernel_launch(void* const* d_in, const int* in_sizes, int n_in,
                              void* d_out, int out_size, void* d_ws, size_t ws_size,
                              hipStream_t stream) {
  (void)in_sizes; (void)n_in; (void)out_size; (void)ws_size;

  const float* x0    = (const float*)d_in[0];
  const int*   ei    = (const int*)d_in[1];
  const float* attr  = (const float*)d_in[2];
  const int*   batch = (const int*)d_in[3];
  const float* ow    = (const float*)d_in[4];
  const float* pa    = (const float*)d_in[5];

  char* ws = (char*)d_ws;
  __bf16* hsw = (__bf16*)ws; ws += (size_t)N_EDGES * 64 * 2;   // 8 MB
  __bf16* pB  = (__bf16*)ws; ws += (size_t)64 * 64 * 64 * 2;   // 512 KB (max K=4096)
  float*  agg = (float*)ws;  ws += (size_t)N_NODES * 64 * 4;   // 4 MB
  float*  R   = (float*)ws;  ws += (size_t)N_NODES * 64 * 4;
  float*  B2x = (float*)ws;  ws += (size_t)N_NODES * 64 * 4;
  float*  xA  = (float*)ws;  ws += (size_t)N_NODES * 64 * 4;
  float*  xB  = (float*)ws;

  const int* src = ei;
  const int* dst = ei + N_EDGES;

  const float* xcur = x0;
  float* bufs[2] = {xA, xB};
  int in_ch = 4;
  for (int L = 0; L < 3; ++L) {
    const float* w1   = (const float*)d_in[6 + 7 * L + 0];
    const float* b1   = (const float*)d_in[6 + 7 * L + 1];
    const float* aL   = (const float*)d_in[6 + 7 * L + 2];
    const float* w2   = (const float*)d_in[6 + 7 * L + 3];
    const float* b2   = (const float*)d_in[6 + 7 * L + 4];
    const float* root = (const float*)d_in[6 + 7 * L + 5];
    const float* bias = (const float*)d_in[6 + 7 * L + 6];

    int totB = in_ch * 64 * 64;
    k_pack_w2<<<(totB + 255) / 256, 256, 0, stream>>>(w2, pB, in_ch);
    k_edge_h<<<(N_EDGES * 64) / 256, 256, 0, stream>>>(attr, w1, b1, aL, hsw);
    k_xroot<<<(N_NODES * 64) / 256, 256, 0, stream>>>(xcur, root, bias, b2, R, B2x, in_ch);
    hipMemsetAsync(agg, 0, (size_t)N_NODES * 64 * 4, stream);
    k_nnconv<<<N_EDGES / 128, 256, 0, stream>>>(xcur, hsw, pB, src, dst, B2x, agg, in_ch);
    float* xn = bufs[L & 1];
    k_finalize<<<(N_NODES * 64) / 256, 256, 0, stream>>>(agg, R, pa, xn);
    xcur = xn;
    in_ch = 64;
  }

  hipMemsetAsync(d_out, 0, (size_t)N_GRAPHS * 4, stream);
  k_pool<<<N_NODES / 256, 256, 0, stream>>>(xcur, batch, ow, (float*)d_out);
}
